// MinGRULayer_18442589569185
// MI455X (gfx1250) — compile-verified, matrix operand order checked by reading, not run on hardware
//
#include <hip/hip_runtime.h>
#include <hip/hip_bf16.h>

// ---------------- problem constants ----------------
#define D_MODEL 1024
#define BATCH   4
#define SEQ     4096
#define MROWS   (BATCH * SEQ)      // 16384 rows of the fused GEMM

// GEMM tiling
#define BM 128
#define BN 128
#define BK 32
#define KSTEPS (D_MODEL / BK)      // 32
#define LSTR 40                    // LDS row stride in ushorts = 80B (64B data + 16B TDM pad)

// scan chunking
#define KC 16                      // chunks per channel
#define LC 256                     // timesteps per chunk (KC*LC == SEQ)

typedef __attribute__((ext_vector_type(16))) __bf16          v16bf;
typedef __attribute__((ext_vector_type(8)))  float           v8f;
typedef __attribute__((ext_vector_type(8)))  unsigned short  ushort8;
typedef __attribute__((ext_vector_type(4)))  unsigned int    u32x4;
typedef __attribute__((ext_vector_type(8)))  int             i32x8;
typedef __attribute__((ext_vector_type(4)))  int             i32x4;

static __device__ __forceinline__ unsigned short f2bf(float f) {
  union { float f; unsigned u; } v; v.f = f;
  unsigned r = v.u + 0x7FFFu + ((v.u >> 16) & 1u);   // round-to-nearest-even
  return (unsigned short)(r >> 16);
}

// ---------------- conversion kernels ----------------
__global__ void cvt_x_bf16(const float* __restrict__ src,
                           unsigned short* __restrict__ dst, int n) {
  int i = blockIdx.x * blockDim.x + threadIdx.x;
  int stride = gridDim.x * blockDim.x;
  for (; i < n; i += stride) dst[i] = f2bf(src[i]);
}

// Convert and transpose both weight matrices: W[k][n] (fp32) -> WT[n][k] (bf16).
__global__ void cvt_w_bf16T(const float* __restrict__ Wz, const float* __restrict__ Wh,
                            unsigned short* __restrict__ WzT, unsigned short* __restrict__ WhT) {
  int tid = blockIdx.x * blockDim.x + threadIdx.x;
  if (tid >= 2 * D_MODEL * D_MODEL) return;
  int mat = tid >> 20;
  int rem = tid & ((1 << 20) - 1);
  int k = rem >> 10;
  int n = rem & (D_MODEL - 1);
  const float* s = mat ? Wh : Wz;
  unsigned short* d = mat ? WhT : WzT;
  d[(size_t)n * D_MODEL + k] = f2bf(s[(size_t)k * D_MODEL + n]);
}

// ---------------- TDM 2D tile load: 128 rows x 32 bf16, row stride 1024 elems ----------------
// D# group0: count=1 | lds_addr | global_addr[56:0] | type=2
// D# group1: data_size=2B, pad_enable, pad_interval=3 (pad after 8x8B=64B),
//            pad_amount=3 (4 dwords = 16B)  -> LDS row stride 80B == LSTR ushorts,
//            tensor_dim0=32, tensor_dim1=128, tile_dim0=32, tile_dim1=128, dim0_stride=1024.
static __device__ __forceinline__ void tdm_load_tile(unsigned lds_off, const void* gptr) {
  unsigned long long ga = (unsigned long long)gptr;
  u32x4 g0;
  g0[0] = 1u;                                               // count=1, user mode, no gather
  g0[1] = lds_off;                                          // lds_addr
  g0[2] = (unsigned)ga;                                     // global_addr[31:0]
  g0[3] = (unsigned)((ga >> 32) & 0x1FFFFFFull) | (2u << 30); // addr[56:32] | type=2
  i32x8 g1;
  g1[0] = (int)((1u << 16) | (1u << 20) | (3u << 22) | (3u << 25)); // dsize=2B, pad en/int/amt
  g1[1] = (int)(32u << 16);          // tensor_dim0 = 32 (low 16 bits at 63:48)
  g1[2] = (int)(128u << 16);         // tensor_dim1 = 128 (low 16 bits at 95:80)
  g1[3] = (int)(32u << 16);          // tile_dim0 = 32 (bits 127:112)
  g1[4] = 128;                       // tile_dim1 = 128 (bits 143:128)
  g1[5] = 1024;                      // tensor_dim0_stride = 1024 elems (bits 191:160)
  g1[6] = 0;
  g1[7] = 0;
  i32x4 z4 = {};
  i32x8 z8 = {};
  __builtin_amdgcn_tensor_load_to_lds(g0, g1, z4, z4, z8, 0);  // 6-arg (clang-23) form
}

// ---------------- fused dual GEMM + gate epilogue ----------------
// 16-bit A-fragment layout (16x32): lanes 0-15 hold row M=lane, K {0..7} in v0-3 and
// {16..23} in v4-7; lanes 16-31 hold K {8..15} / {24..31}. B-fragment mirrors this with
// lane = output column; W stored transposed [n][k] so both loads are contiguous in K.
struct BfPair { ushort8 lo, hi; };

static __device__ __forceinline__ v16bf load_frag(const unsigned short* lds, int row, int koff) {
  BfPair p;
  p.lo = *(const ushort8*)(lds + row * LSTR + koff);        // K block 0..7 (or 8..15)
  p.hi = *(const ushort8*)(lds + row * LSTR + 16 + koff);   // K block 16..23 (or 24..31)
  return __builtin_bit_cast(v16bf, p);
}

__global__ __launch_bounds__(256)
void gemm_gates(const unsigned short* __restrict__ xbf,
                const unsigned short* __restrict__ WzT,
                const unsigned short* __restrict__ WhT,
                const float* __restrict__ bz, const float* __restrict__ bh,
                float* __restrict__ aout,      // (1 - z)      -> workspace
                float* __restrict__ bout) {    // z * h_tilde  -> d_out (scanned in place later)
  __shared__ unsigned short sX[2][BM * LSTR];
  __shared__ unsigned short sZ[2][BN * LSTR];
  __shared__ unsigned short sH[2][BN * LSTR];

  const int t    = threadIdx.x;
  const int lane = t & 31;
  const int wv   = t >> 5;           // 8 waves
  const int wrow = wv >> 2;          // 2 wave-rows  (64 M each)
  const int wcol = wv & 3;           // 4 wave-cols  (32 N each)
  const int lrow = lane & 15;
  const int koff = (lane >> 4) * 8;

  const int m0 = blockIdx.y * BM;
  const int n0 = blockIdx.x * BN;

  v8f accz[4][2], acch[4][2];
  #pragma unroll
  for (int i = 0; i < 4; ++i)
    #pragma unroll
    for (int j = 0; j < 2; ++j) { accz[i][j] = {}; acch[i][j] = {}; }

  // wave 0 drives the Tensor Data Mover; TENSORcnt is per-wave.
  if (wv == 0) {
    tdm_load_tile((unsigned)(size_t)&sX[0][0], xbf + (size_t)m0 * D_MODEL + 0);
    tdm_load_tile((unsigned)(size_t)&sZ[0][0], WzT + (size_t)n0 * D_MODEL + 0);
    tdm_load_tile((unsigned)(size_t)&sH[0][0], WhT + (size_t)n0 * D_MODEL + 0);
  }

  for (int kt = 0; kt < KSTEPS; ++kt) {
    if (wv == 0) {
      if (kt + 1 < KSTEPS) {
        const int kn = (kt + 1) * BK;
        const int nb = (kt + 1) & 1;
        tdm_load_tile((unsigned)(size_t)&sX[nb][0], xbf + (size_t)m0 * D_MODEL + kn);
        tdm_load_tile((unsigned)(size_t)&sZ[nb][0], WzT + (size_t)n0 * D_MODEL + kn);
        tdm_load_tile((unsigned)(size_t)&sH[nb][0], WhT + (size_t)n0 * D_MODEL + kn);
        __builtin_amdgcn_s_wait_tensorcnt(3);   // oldest 3 (current tile) complete
      } else {
        __builtin_amdgcn_s_wait_tensorcnt(0);
      }
    }
    __syncthreads();                            // publish LDS tile to all waves

    const unsigned short* bX = &sX[kt & 1][0];
    const unsigned short* bZ = &sZ[kt & 1][0];
    const unsigned short* bH = &sH[kt & 1][0];

    v16bf af[4], bzf[2], bhf[2];
    #pragma unroll
    for (int i = 0; i < 4; ++i)
      af[i] = load_frag(bX, wrow * 64 + i * 16 + lrow, koff);
    #pragma unroll
    for (int j = 0; j < 2; ++j) {
      bzf[j] = load_frag(bZ, wcol * 32 + j * 16 + lrow, koff);
      bhf[j] = load_frag(bH, wcol * 32 + j * 16 + lrow, koff);
    }

    #pragma unroll
    for (int i = 0; i < 4; ++i)
      #pragma unroll
      for (int j = 0; j < 2; ++j) {
        accz[i][j] = __builtin_amdgcn_wmma_f32_16x16x32_bf16(
            false, af[i], false, bzf[j], (short)0, accz[i][j], false, false);
        acch[i][j] = __builtin_amdgcn_wmma_f32_16x16x32_bf16(
            false, af[i], false, bhf[j], (short)0, acch[i][j], false, false);
      }
    __syncthreads();                            // done reading this buffer; TDM may refill it
  }

  // C/D layout: VGPR r, lanes 0-15 -> (M=r, N=lane); lanes 16-31 -> (M=8+r, N=lane-16)
  const int mloff = (lane >> 4) * 8;
  const int ncol  = lane & 15;
  #pragma unroll
  for (int i = 0; i < 4; ++i) {
    #pragma unroll
    for (int j = 0; j < 2; ++j) {
      const int n = n0 + wcol * 32 + j * 16 + ncol;
      const float bzv = bz[n], bhv = bh[n];
      #pragma unroll
      for (int r = 0; r < 8; ++r) {
        const int m = m0 + wrow * 64 + i * 16 + mloff + r;
        const float pz = accz[i][j][r] + bzv;
        const float ph = acch[i][j][r] + bhv;
        const float zv = __builtin_amdgcn_rcpf(1.0f + __expf(-pz));  // v_rcp_f32 sigmoid
        const size_t idx = (size_t)m * D_MODEL + n;
        aout[idx] = 1.0f - zv;     // a_t
        bout[idx] = zv * ph;       // b_t
      }
    }
  }
}

// ---------------- chunked linear recurrence  h_t = a_t * h_{t-1} + b_t ----------------
__global__ void scan_reduce(const float* __restrict__ a, const float* __restrict__ b,
                            float* __restrict__ P, float* __restrict__ Q) {
  const int tid   = blockIdx.x * blockDim.x + threadIdx.x;      // 65536
  const int d     = tid & (D_MODEL - 1);
  const int k     = (tid >> 10) & (KC - 1);
  const int batch = tid >> 14;
  size_t base = (size_t)batch * SEQ * D_MODEL + (size_t)k * LC * D_MODEL + d;
  float p = 1.0f, q = 0.0f;
  for (int s = 0; s < LC; ++s) {
    const float av = a[base + (size_t)s * D_MODEL];
    const float bv = b[base + (size_t)s * D_MODEL];
    q = fmaf(av, q, bv);
    p *= av;
  }
  P[tid] = p;
  Q[tid] = q;
}

__global__ void scan_prefix(const float* __restrict__ P, const float* __restrict__ Q,
                            float* __restrict__ Hin) {
  const int tid   = blockIdx.x * blockDim.x + threadIdx.x;      // 4096 channels
  const int d     = tid & (D_MODEL - 1);
  const int batch = tid >> 10;
  float h = 0.0f;
  for (int k = 0; k < KC; ++k) {
    const int idx = (batch * KC + k) * D_MODEL + d;
    Hin[idx] = h;                      // exclusive: state entering chunk k
    h = fmaf(P[idx], h, Q[idx]);
  }
}

__global__ void scan_apply(const float* __restrict__ a, float* __restrict__ hb,
                           const float* __restrict__ Hin) {
  const int tid   = blockIdx.x * blockDim.x + threadIdx.x;      // 65536
  const int d     = tid & (D_MODEL - 1);
  const int k     = (tid >> 10) & (KC - 1);
  const int batch = tid >> 14;
  size_t base = (size_t)batch * SEQ * D_MODEL + (size_t)k * LC * D_MODEL + d;
  float h = Hin[tid];
  for (int s = 0; s < LC; ++s) {
    const size_t idx = base + (size_t)s * D_MODEL;
    h = fmaf(a[idx], h, hb[idx]);      // read b_t, emit h_t in place
    hb[idx] = h;
  }
}

// ---------------- launcher ----------------
extern "C" void kernel_launch(void* const* d_in, const int* in_sizes, int n_in,
                              void* d_out, int out_size, void* d_ws, size_t ws_size,
                              hipStream_t stream) {
  const float* x  = (const float*)d_in[0];
  const float* Wz = (const float*)d_in[1];
  const float* bz = (const float*)d_in[2];
  const float* Wh = (const float*)d_in[3];
  const float* bh = (const float*)d_in[4];
  float* bout = (float*)d_out;

  char* ws = (char*)d_ws;
  size_t off = 0;
  unsigned short* xbf = (unsigned short*)(ws + off); off += (size_t)MROWS * D_MODEL * 2;   // 33.5 MB
  unsigned short* WzT = (unsigned short*)(ws + off); off += (size_t)D_MODEL * D_MODEL * 2; //  2.1 MB
  unsigned short* WhT = (unsigned short*)(ws + off); off += (size_t)D_MODEL * D_MODEL * 2; //  2.1 MB
  float* aout = (float*)(ws + off);                  off += (size_t)MROWS * D_MODEL * 4;   // 64  MB
  float* P    = (float*)(ws + off);                  off += (size_t)BATCH * KC * D_MODEL * 4;
  float* Q    = (float*)(ws + off);                  off += (size_t)BATCH * KC * D_MODEL * 4;
  float* Hin  = (float*)(ws + off);                  off += (size_t)BATCH * KC * D_MODEL * 4;
  (void)ws_size; (void)in_sizes; (void)n_in; (void)out_size;

  cvt_x_bf16<<<4096, 256, 0, stream>>>(x, xbf, MROWS * D_MODEL);
  cvt_w_bf16T<<<(2 * D_MODEL * D_MODEL) / 256, 256, 0, stream>>>(Wz, Wh, WzT, WhT);

  dim3 grid(D_MODEL / BN, MROWS / BM);   // (8, 128)
  gemm_gates<<<grid, 256, 0, stream>>>(xbf, WzT, WhT, bz, bh, aout, bout);

  scan_reduce<<<(BATCH * KC * D_MODEL) / 256, 256, 0, stream>>>(aout, bout, P, Q);
  scan_prefix<<<(BATCH * D_MODEL) / 256, 256, 0, stream>>>(P, Q, Hin);
  scan_apply<<<(BATCH * KC * D_MODEL) / 256, 256, 0, stream>>>(aout, bout, Hin);
}
